// EGNNLayer_7541962572405
// MI455X (gfx1250) — matrix operand order repaired and checked by reading, not compiled, not run on hardware
//
#include <hip/hip_runtime.h>
#include <hip/hip_bf16.h>

// ---------------------------------------------------------------------------
// EGNN layer for MI455X (gfx1250), bf16 WMMA with f32 accumulate.
// ---------------------------------------------------------------------------

typedef __attribute__((ext_vector_type(16))) __bf16 v16bf;
typedef __attribute__((ext_vector_type(8)))  float  v8f;

union FragAB { v16bf v; uint4 q[2]; };

__device__ __forceinline__ unsigned short f2bf(float f) {
    unsigned u = __float_as_uint(f);
    u += 0x7FFFu + ((u >> 16) & 1u);      // round-to-nearest-even
    return (unsigned short)(u >> 16);
}

__device__ __forceinline__ float silu_f(float v) {
    return v / (1.0f + __expf(-v));
}

#define WMMA_BF16(A, B, C) \
    __builtin_amdgcn_wmma_f32_16x16x32_bf16(false, (A), false, (B), (short)0, (C), false, false)

// ---------------------------------------------------------------------------
// Weight prep: f32 [K][128] -> bf16 transposed [128][Kpad]
// ---------------------------------------------------------------------------
__global__ void egnn_prep(const float* __restrict__ Wm1, const float* __restrict__ Wm2,
                          const float* __restrict__ Wc1, const float* __restrict__ Wn1,
                          const float* __restrict__ Wn2, const float* __restrict__ Wc2,
                          unsigned short* __restrict__ Wm1t, unsigned short* __restrict__ Wm2t,
                          unsigned short* __restrict__ Wc1t, unsigned short* __restrict__ Wn1t,
                          unsigned short* __restrict__ Wn2t, float* __restrict__ Wc2f) {
    int i = blockIdx.x * blockDim.x + threadIdx.x;
    const int T0 = 128 * 288;   // Wm1t  (K=260 padded to 288)
    const int T1 = 128 * 128;   // Wm2t / Wc1t / Wn2t
    const int T3 = 128 * 256;   // Wn1t
    if (i < T0) {
        int n = i / 288, k = i % 288;
        Wm1t[i] = (k < 260) ? f2bf(Wm1[k * 128 + n]) : (unsigned short)0;
        return;
    }
    i -= T0;
    if (i < T1) { int n = i / 128, k = i % 128; Wm2t[i] = f2bf(Wm2[k * 128 + n]); return; }
    i -= T1;
    if (i < T1) { int n = i / 128, k = i % 128; Wc1t[i] = f2bf(Wc1[k * 128 + n]); return; }
    i -= T1;
    if (i < T3) { int n = i / 256, k = i % 256; Wn1t[i] = f2bf(Wn1[k * 128 + n]); return; }
    i -= T3;
    if (i < T1) { int n = i / 128, k = i % 128; Wn2t[i] = f2bf(Wn2[k * 128 + n]); return; }
    i -= T1;
    if (i < 128) { Wc2f[i] = Wc2[i]; }
}

// ---------------------------------------------------------------------------
// Edge kernel: fused message MLP + coord head + atomic aggregation.
// 128 threads (4 waves), 16 edges per block. Each wave owns 32 output cols.
// ---------------------------------------------------------------------------
__global__ __launch_bounds__(128) void egnn_edge(
    const float* __restrict__ x, const float* __restrict__ pos,
    const int* __restrict__ ei, const float* __restrict__ ea,
    const unsigned short* __restrict__ Wm1t, const float* __restrict__ bm1,
    const unsigned short* __restrict__ Wm2t, const float* __restrict__ bm2,
    const unsigned short* __restrict__ Wc1t, const float* __restrict__ bc1,
    const float* __restrict__ Wc2f, const float* __restrict__ bc2,
    float* __restrict__ agg, float* __restrict__ posOut, int E) {

    __shared__ alignas(16) unsigned short sIn[16][296];  // msg_input bf16 (288 used)
    __shared__ alignas(16) unsigned short sH[16][136];   // hidden bf16 (128 used)
    __shared__ alignas(16) unsigned short sM[16][136];   // msg bf16 (128 used)
    __shared__ float sDiff[16][3];
    __shared__ int   sRow[16];
    __shared__ float sCW[16];

    const int tid  = threadIdx.x;
    const int wave = tid >> 5;
    const int lane = tid & 31;
    const int half = lane >> 4;
    const int ln   = lane & 15;
    const int e0   = blockIdx.x * 16;

    // ---- stage msg_input = [x[row] | x[col] | dist | edge_attr | 0-pad] ----
    #pragma unroll
    for (int i = 0; i < 4; ++i) {
        const int el = wave * 4 + i;
        const int e  = e0 + el;
        const int r  = ei[e];
        const int c  = ei[E + e];
        const float4 xr = *(const float4*)&x[(size_t)r * 128 + lane * 4];
        const float4 xc = *(const float4*)&x[(size_t)c * 128 + lane * 4];
        unsigned short* d0 = &sIn[el][lane * 4];
        d0[0] = f2bf(xr.x); d0[1] = f2bf(xr.y); d0[2] = f2bf(xr.z); d0[3] = f2bf(xr.w);
        unsigned short* d1 = &sIn[el][128 + lane * 4];
        d1[0] = f2bf(xc.x); d1[1] = f2bf(xc.y); d1[2] = f2bf(xc.z); d1[3] = f2bf(xc.w);
        if (lane == 0) {
            const float dx = pos[r * 3 + 0] - pos[c * 3 + 0];
            const float dy = pos[r * 3 + 1] - pos[c * 3 + 1];
            const float dz = pos[r * 3 + 2] - pos[c * 3 + 2];
            sDiff[el][0] = dx; sDiff[el][1] = dy; sDiff[el][2] = dz;
            sRow[el] = r;
            sIn[el][256] = f2bf(sqrtf(dx * dx + dy * dy + dz * dz));
            sIn[el][257] = f2bf(ea[e * 3 + 0]);
            sIn[el][258] = f2bf(ea[e * 3 + 1]);
            sIn[el][259] = f2bf(ea[e * 3 + 2]);
        }
        if (lane >= 4) sIn[el][256 + lane] = 0;  // zero K = 260..287
    }
    if (tid < 16) sCW[tid] = bc2[0];
    __syncthreads();

    const int n0 = wave * 32;
    const int nA = n0 + ln;
    const int nB = n0 + 16 + ln;

    // ---- GEMM1: h = silu(msg_input @ Wm1 + bm1), K = 288 ----
    v8f h0 = {}, h1 = {};
    #pragma unroll
    for (int kc = 0; kc < 9; ++kc) {
        FragAB A, B0, B1;
        const int ka = kc * 32 + half * 16;
        A.q[0] = *(const uint4*)&sIn[ln][ka];
        A.q[1] = *(const uint4*)&sIn[ln][ka + 8];
        const int kb = kc * 32 + half * 16;
        const unsigned short* bp0 = &Wm1t[nA * 288 + kb];
        const unsigned short* bp1 = &Wm1t[nB * 288 + kb];
        B0.q[0] = *(const uint4*)bp0; B0.q[1] = *(const uint4*)(bp0 + 8);
        B1.q[0] = *(const uint4*)bp1; B1.q[1] = *(const uint4*)(bp1 + 8);
        h0 = WMMA_BF16(A.v, B0.v, h0);
        h1 = WMMA_BF16(A.v, B1.v, h1);
    }
    {
        const float ba = bm1[nA], bb = bm1[nB];
        #pragma unroll
        for (int r = 0; r < 8; ++r) {
            const int m = r + half * 8;
            sH[m][nA] = f2bf(silu_f(h0[r] + ba));
            sH[m][nB] = f2bf(silu_f(h1[r] + bb));
        }
    }
    __syncthreads();

    // ---- GEMM2: msg = silu(h @ Wm2 + bm2), K = 128; atomic segment-sum ----
    v8f g0 = {}, g1 = {};
    #pragma unroll
    for (int kc = 0; kc < 4; ++kc) {
        FragAB A, B0, B1;
        const int ka = kc * 32 + half * 16;
        A.q[0] = *(const uint4*)&sH[ln][ka];
        A.q[1] = *(const uint4*)&sH[ln][ka + 8];
        const int kb = kc * 32 + half * 16;
        const unsigned short* bp0 = &Wm2t[nA * 128 + kb];
        const unsigned short* bp1 = &Wm2t[nB * 128 + kb];
        B0.q[0] = *(const uint4*)bp0; B0.q[1] = *(const uint4*)(bp0 + 8);
        B1.q[0] = *(const uint4*)bp1; B1.q[1] = *(const uint4*)(bp1 + 8);
        g0 = WMMA_BF16(A.v, B0.v, g0);
        g1 = WMMA_BF16(A.v, B1.v, g1);
    }
    {
        const float ba = bm2[nA], bb = bm2[nB];
        #pragma unroll
        for (int r = 0; r < 8; ++r) {
            const int m = r + half * 8;
            const float v0 = silu_f(g0[r] + ba);
            const float v1 = silu_f(g1[r] + bb);
            sM[m][nA] = f2bf(v0);
            sM[m][nB] = f2bf(v1);
            const size_t base = (size_t)sRow[m] * 128;
            atomicAdd(&agg[base + nA], v0);
            atomicAdd(&agg[base + nB], v1);
        }
    }
    __syncthreads();

    // ---- GEMM3: c1 = silu(msg @ Wc1 + bc1); coord_w = c1 @ Wc2 + bc2 ----
    v8f c0 = {}, c1 = {};
    #pragma unroll
    for (int kc = 0; kc < 4; ++kc) {
        FragAB A, B0, B1;
        const int ka = kc * 32 + half * 16;
        A.q[0] = *(const uint4*)&sM[ln][ka];
        A.q[1] = *(const uint4*)&sM[ln][ka + 8];
        const int kb = kc * 32 + half * 16;
        const unsigned short* bp0 = &Wc1t[nA * 128 + kb];
        const unsigned short* bp1 = &Wc1t[nB * 128 + kb];
        B0.q[0] = *(const uint4*)bp0; B0.q[1] = *(const uint4*)(bp0 + 8);
        B1.q[0] = *(const uint4*)bp1; B1.q[1] = *(const uint4*)(bp1 + 8);
        c0 = WMMA_BF16(A.v, B0.v, c0);
        c1 = WMMA_BF16(A.v, B1.v, c1);
    }
    {
        const float ba = bc1[nA], bb = bc1[nB];
        const float wa = Wc2f[nA], wb = Wc2f[nB];
        float p[8];
        #pragma unroll
        for (int r = 0; r < 8; ++r)
            p[r] = silu_f(c0[r] + ba) * wa + silu_f(c1[r] + bb) * wb;
        // reduce over the 16 lanes of each half-wave (same m, different n)
        #pragma unroll
        for (int mask = 1; mask < 16; mask <<= 1) {
            #pragma unroll
            for (int r = 0; r < 8; ++r)
                p[r] += __shfl_xor(p[r], mask, 32);
        }
        if (ln == 0) {
            #pragma unroll
            for (int r = 0; r < 8; ++r)
                atomicAdd(&sCW[r + half * 8], p[r]);
        }
    }
    __syncthreads();

    // ---- pos_new += diff * coord_w (atomic scatter) ----
    if (tid < 48) {
        const int el = tid / 3, ax = tid % 3;
        atomicAdd(&posOut[(size_t)sRow[el] * 3 + ax], sDiff[el][ax] * sCW[el]);
    }
}

// ---------------------------------------------------------------------------
// Node kernel: x_new = silu(concat[x, agg] @ Wn1 + bn1) @ Wn2 + bn2
// ---------------------------------------------------------------------------
__global__ __launch_bounds__(128) void egnn_node(
    const float* __restrict__ x, const float* __restrict__ agg,
    const unsigned short* __restrict__ Wn1t, const float* __restrict__ bn1,
    const unsigned short* __restrict__ Wn2t, const float* __restrict__ bn2,
    float* __restrict__ xOut) {

    __shared__ alignas(16) unsigned short sIn[16][264];  // concat bf16 (256 used)
    __shared__ alignas(16) unsigned short sH[16][136];

    const int tid  = threadIdx.x;
    const int wave = tid >> 5;
    const int lane = tid & 31;
    const int half = lane >> 4;
    const int ln   = lane & 15;
    const int g0b  = blockIdx.x * 16;

    #pragma unroll
    for (int i = 0; i < 4; ++i) {
        const int nl = wave * 4 + i;
        const size_t g = (size_t)(g0b + nl);
        const float4 xv = *(const float4*)&x[g * 128 + lane * 4];
        const float4 av = *(const float4*)&agg[g * 128 + lane * 4];
        unsigned short* d0 = &sIn[nl][lane * 4];
        d0[0] = f2bf(xv.x); d0[1] = f2bf(xv.y); d0[2] = f2bf(xv.z); d0[3] = f2bf(xv.w);
        unsigned short* d1 = &sIn[nl][128 + lane * 4];
        d1[0] = f2bf(av.x); d1[1] = f2bf(av.y); d1[2] = f2bf(av.z); d1[3] = f2bf(av.w);
    }
    __syncthreads();

    const int n0 = wave * 32;
    const int nA = n0 + ln;
    const int nB = n0 + 16 + ln;

    v8f h0 = {}, h1 = {};
    #pragma unroll
    for (int kc = 0; kc < 8; ++kc) {
        FragAB A, B0, B1;
        const int ka = kc * 32 + half * 16;
        A.q[0] = *(const uint4*)&sIn[ln][ka];
        A.q[1] = *(const uint4*)&sIn[ln][ka + 8];
        const int kb = kc * 32 + half * 16;
        const unsigned short* bp0 = &Wn1t[nA * 256 + kb];
        const unsigned short* bp1 = &Wn1t[nB * 256 + kb];
        B0.q[0] = *(const uint4*)bp0; B0.q[1] = *(const uint4*)(bp0 + 8);
        B1.q[0] = *(const uint4*)bp1; B1.q[1] = *(const uint4*)(bp1 + 8);
        h0 = WMMA_BF16(A.v, B0.v, h0);
        h1 = WMMA_BF16(A.v, B1.v, h1);
    }
    {
        const float ba = bn1[nA], bb = bn1[nB];
        #pragma unroll
        for (int r = 0; r < 8; ++r) {
            const int m = r + half * 8;
            sH[m][nA] = f2bf(silu_f(h0[r] + ba));
            sH[m][nB] = f2bf(silu_f(h1[r] + bb));
        }
    }
    __syncthreads();

    v8f o0 = {}, o1 = {};
    #pragma unroll
    for (int kc = 0; kc < 4; ++kc) {
        FragAB A, B0, B1;
        const int ka = kc * 32 + half * 16;
        A.q[0] = *(const uint4*)&sH[ln][ka];
        A.q[1] = *(const uint4*)&sH[ln][ka + 8];
        const int kb = kc * 32 + half * 16;
        const unsigned short* bp0 = &Wn2t[nA * 128 + kb];
        const unsigned short* bp1 = &Wn2t[nB * 128 + kb];
        B0.q[0] = *(const uint4*)bp0; B0.q[1] = *(const uint4*)(bp0 + 8);
        B1.q[0] = *(const uint4*)bp1; B1.q[1] = *(const uint4*)(bp1 + 8);
        o0 = WMMA_BF16(A.v, B0.v, o0);
        o1 = WMMA_BF16(A.v, B1.v, o1);
    }
    {
        const float ba = bn2[nA], bb = bn2[nB];
        #pragma unroll
        for (int r = 0; r < 8; ++r) {
            const int m = r + half * 8;
            const size_t base = (size_t)(g0b + m) * 128;
            xOut[base + nA] = o0[r] + ba;
            xOut[base + nB] = o1[r] + bb;
        }
    }
}

// ---------------------------------------------------------------------------
// Host launcher
// ---------------------------------------------------------------------------
extern "C" void kernel_launch(void* const* d_in, const int* in_sizes, int n_in,
                              void* d_out, int out_size, void* d_ws, size_t ws_size,
                              hipStream_t stream) {
    constexpr int N = 50000, E = 640000;

    const float* x   = (const float*)d_in[0];
    const float* pos = (const float*)d_in[1];
    const int*   ei  = (const int*)d_in[2];
    const float* ea  = (const float*)d_in[3];
    const float* Wm1 = (const float*)d_in[4];  const float* bm1 = (const float*)d_in[5];
    const float* Wm2 = (const float*)d_in[6];  const float* bm2 = (const float*)d_in[7];
    const float* Wn1 = (const float*)d_in[8];  const float* bn1 = (const float*)d_in[9];
    const float* Wn2 = (const float*)d_in[10]; const float* bn2 = (const float*)d_in[11];
    const float* Wc1 = (const float*)d_in[12]; const float* bc1 = (const float*)d_in[13];
    const float* Wc2 = (const float*)d_in[14]; const float* bc2 = (const float*)d_in[15];

    char* ws = (char*)d_ws;
    size_t off = 0;
    float* agg = (float*)(ws + off);                 off += (size_t)N * 128 * 4;  // 25.6 MB
    unsigned short* Wm1t = (unsigned short*)(ws + off); off += 128 * 288 * 2;
    unsigned short* Wm2t = (unsigned short*)(ws + off); off += 128 * 128 * 2;
    unsigned short* Wc1t = (unsigned short*)(ws + off); off += 128 * 128 * 2;
    unsigned short* Wn1t = (unsigned short*)(ws + off); off += 128 * 256 * 2;
    unsigned short* Wn2t = (unsigned short*)(ws + off); off += 128 * 128 * 2;
    float* Wc2f = (float*)(ws + off);                off += 128 * 4;

    float* xOut   = (float*)d_out;
    float* posOut = xOut + (size_t)N * 128;

    // zero the aggregation buffer; seed pos_new with pos (edge kernel adds onto it)
    hipMemsetAsync(agg, 0, (size_t)N * 128 * 4, stream);
    hipMemcpyAsync(posOut, pos, (size_t)N * 3 * 4, hipMemcpyDeviceToDevice, stream);

    const int prep_total = 128 * 288 + 3 * 128 * 128 + 128 * 256 + 128;
    egnn_prep<<<(prep_total + 255) / 256, 256, 0, stream>>>(
        Wm1, Wm2, Wc1, Wn1, Wn2, Wc2, Wm1t, Wm2t, Wc1t, Wn1t, Wn2t, Wc2f);

    egnn_edge<<<E / 16, 128, 0, stream>>>(
        x, pos, ei, ea, Wm1t, bm1, Wm2t, bm2, Wc1t, bc1, Wc2f, bc2, agg, posOut, E);

    egnn_node<<<N / 16, 128, 0, stream>>>(
        x, agg, Wn1t, bn1, Wn2t, bn2, xOut);
}
